// ScaledDotProductAttention_87393994539166
// MI455X (gfx1250) — matrix operand order repaired and checked by reading, not compiled, not run on hardware
//
#include <hip/hip_runtime.h>
#include <hip/hip_bf16.h>

typedef __attribute__((ext_vector_type(16))) _Float16 v16h;
typedef __attribute__((ext_vector_type(8)))  float    v8f;
typedef __attribute__((ext_vector_type(4)))  float    v4f;

#define BATCH 8
#define SEQ   4096
#define DIM   64
#define TQ    16            // q rows per workgroup
#define NW    8             // waves per workgroup (wave32)
#define NT    (NW * 32)     // 256 threads
#define SP    (SEQ + 4)     // padded LDS row stride (floats); 4100%4==0 keeps b128 alignment

// A-fragment (16-bit, 16x32 MxK): lane holds row m = lane%16.
// element e -> K: lanes 0-15: K = e (e<8), e+8 (e>=8); lanes 16-31: +8.
__device__ __forceinline__ int aKmap(int e, int hi) {
    return e + 8 * hi + ((e >> 3) << 3);
}
// B-fragment (16-bit, 32x16 KxN): lane holds col n = lane%16.
// element e -> K (this code's contiguous order, as packA): lanes 0-15: K = e; lanes 16-31: K = e + 16.
__device__ __forceinline__ int bKmap(int e, int hi) {
    return e + 16 * hi;
}

// Pack 16 f16 A-fragment elements from 16 contiguous f32:
// elements 0..15 <- base[16*hi .. 16*hi+15]
__device__ __forceinline__ v16h packA(const float* base, int hi) {
    v4f p0 = *(const v4f*)(base + 16 * hi);
    v4f p1 = *(const v4f*)(base + 16 * hi + 4);
    v4f p2 = *(const v4f*)(base + 16 * hi + 8);
    v4f p3 = *(const v4f*)(base + 16 * hi + 12);
    v16h a;
#pragma unroll
    for (int j = 0; j < 4; ++j) {
        a[j]      = (_Float16)p0[j];
        a[4 + j]  = (_Float16)p1[j];
        a[8 + j]  = (_Float16)p2[j];
        a[12 + j] = (_Float16)p3[j];
    }
    return a;
}

// Pack 16 f16 B-fragment elements from 16 contiguous f32: base[16*hi .. 16*hi+15]
__device__ __forceinline__ v16h packB(const float* base, int hi) {
    v4f p0 = *(const v4f*)(base + 16 * hi);
    v4f p1 = *(const v4f*)(base + 16 * hi + 4);
    v4f p2 = *(const v4f*)(base + 16 * hi + 8);
    v4f p3 = *(const v4f*)(base + 16 * hi + 12);
    v16h b;
#pragma unroll
    for (int j = 0; j < 4; ++j) {
        b[j]      = (_Float16)p0[j];
        b[4 + j]  = (_Float16)p1[j];
        b[8 + j]  = (_Float16)p2[j];
        b[12 + j] = (_Float16)p3[j];
    }
    return b;
}

__global__ __launch_bounds__(NT) void
attn_fused_kernel(const float* __restrict__ Q,
                  const float* __restrict__ Kmat,
                  const float* __restrict__ V,
                  float* __restrict__ Out,
                  float* __restrict__ Attn)
{
    __shared__ float sS[TQ * SP];        // ~262.4 KB scores -> probabilities
    __shared__ float sQ[TQ * DIM];       // 4 KB staged (pre-scaled) Q tile
    __shared__ float sRed[TQ * 16];      // 1 KB row-reduction scratch
    __shared__ float sStat[TQ * 2];      // row max, 1/rowsum
    __shared__ float sAcc[NW * TQ * DIM];// 32 KB cross-wave PV partials

    const int tid  = threadIdx.x;
    const int lane = tid & 31;
    const int wave = tid >> 5;
    const int hi   = (lane >> 4) & 1;
    const int l16  = lane & 15;

    const int qt    = blockIdx.x % (SEQ / TQ);
    const int batch = blockIdx.x / (SEQ / TQ);
    const int qbase = qt * TQ;

    const float* Qb = Q    + (size_t)batch * SEQ * DIM;
    const float* Kb = Kmat + (size_t)batch * SEQ * DIM;
    const float* Vb = V    + (size_t)batch * SEQ * DIM;

    // ---- stage Q tile to LDS (b128), pre-scaled by 1/sqrt(64) ----
    {
        const v4f* qv = (const v4f*)(Qb + (size_t)qbase * DIM);
        v4f* sqv = (v4f*)sQ;
        for (int i = tid; i < TQ * DIM / 4; i += NT) {
            v4f x = qv[i];
            sqv[i] = x * 0.125f;
        }
    }
    __syncthreads();

    // ---- build per-lane Q A-fragments (d = 0..31 and 32..63) ----
    const v16h aq0 = packA(&sQ[l16 * DIM], hi);
    const v16h aq1 = packA(&sQ[l16 * DIM + 32], hi);

    // ---- phase 1: scores = (Q K^T)/8 with causal mask, into LDS ----
    const int nkt = qt + 1;                 // number of 16-wide key tiles needed
    for (int kt = wave; kt < nkt; kt += NW) {
        const float* Kp = Kb + (size_t)(kt * TQ + l16) * DIM;  // this lane's key row (n)
        if (kt + NW < nkt)                                      // prefetch next tile's row
            __builtin_prefetch(Kb + (size_t)((kt + NW) * TQ + l16) * DIM, 0, 3);

        v16h b0 = packB(Kp, hi);
        v16h b1 = packB(Kp + 32, hi);

        v8f c = {};
        c = __builtin_amdgcn_wmma_f32_16x16x32_f16(false, aq0, false, b0, (short)0, c, false, false);
        c = __builtin_amdgcn_wmma_f32_16x16x32_f16(false, aq1, false, b1, (short)0, c, false, false);

        const int ncol = kt * TQ + l16;
#pragma unroll
        for (int r = 0; r < 8; ++r) {
            int row = r + 8 * hi;           // m within tile
            float s = c[r];
            if (ncol > qbase + row) s = -__builtin_inff();  // causal mask
            sS[row * SP + ncol] = s;
        }
    }
    __syncthreads();

    // ---- phase 2: row softmax stats (16 threads per row, b128 LDS traffic) ----
    const int Kc    = qbase + TQ;           // computed (tile-padded) column count
    const int KcPad = (Kc + 31) & ~31;      // pad to 32 for PV chunks
    const int rr = tid >> 4;
    const int cc = tid & 15;
    v4f* rowv = (v4f*)&sS[rr * SP];         // row stride 4100 floats -> 16B aligned
    const int nchunk = Kc >> 2;             // multiple of 4

    float mx = -__builtin_inff();
    for (int i = cc; i < nchunk; i += 16) {
        v4f x = rowv[i];
        mx = fmaxf(mx, fmaxf(fmaxf(x[0], x[1]), fmaxf(x[2], x[3])));
    }
    sRed[rr * 16 + cc] = mx;
    __syncthreads();
    if (cc == 0) {
        float m2 = -__builtin_inff();
#pragma unroll
        for (int j = 0; j < 16; ++j) m2 = fmaxf(m2, sRed[rr * 16 + j]);
        sStat[rr] = m2;
    }
    __syncthreads();
    const float rmax = sStat[rr];
    float sum = 0.0f;
    for (int i = cc; i < nchunk; i += 16) {
        v4f x = rowv[i];
        v4f p;
#pragma unroll
        for (int j = 0; j < 4; ++j) {
            p[j] = __expf(x[j] - rmax);     // exp(-inf)=0 handles mask
            sum += p[j];
        }
        rowv[i] = p;
    }
    sRed[rr * 16 + cc] = sum;
    __syncthreads();
    if (cc == 0) {
        float s2 = 0.0f;
#pragma unroll
        for (int j = 0; j < 16; ++j) s2 += sRed[rr * 16 + j];
        sStat[TQ + rr] = 1.0f / s2;
    }
    __syncthreads();

    // ---- phase 3: normalize, stream attention to global (b128), zero-fill tail ----
    float* attnBase = Attn + ((size_t)batch * SEQ + qbase) * SEQ;
    for (int r = 0; r < TQ; ++r) {
        const float inv = sStat[TQ + r];
        v4f* arow = (v4f*)(attnBase + (size_t)r * SEQ);
        v4f* srow = (v4f*)&sS[r * SP];
        const int live = Kc >> 2;           // chunks fully inside causal region
        const int pad  = KcPad >> 2;
        for (int i = tid; i < SEQ / 4; i += NT) {
            v4f v;
            if (i < live) {
                v = srow[i] * inv;
                srow[i] = v;                // keep normalized probs for PV
            } else {
                v = (v4f){0.0f, 0.0f, 0.0f, 0.0f};
                if (i < pad) srow[i] = v;   // zero-pad partial PV chunk
            }
            arow[i] = v;
        }
    }
    __syncthreads();

    // ---- phase 4: O = P @ V via WMMA, waves split K in 32-chunks ----
    v8f acc0 = {}, acc1 = {}, acc2 = {}, acc3 = {};
    const int nck = KcPad / 32;
    for (int ck = wave; ck < nck; ck += NW) {
        v16h pa = packA(&sS[l16 * SP + ck * 32], hi);
#pragma unroll
        for (int nt = 0; nt < 4; ++nt) {
            v16h bv;
#pragma unroll
            for (int e = 0; e < 16; ++e) {
                int kk = bKmap(e, hi);
                bv[e] = (_Float16)Vb[(size_t)(ck * 32 + kk) * DIM + nt * 16 + l16];
            }
            if (nt == 0) acc0 = __builtin_amdgcn_wmma_f32_16x16x32_f16(false, pa, false, bv, (short)0, acc0, false, false);
            if (nt == 1) acc1 = __builtin_amdgcn_wmma_f32_16x16x32_f16(false, pa, false, bv, (short)0, acc1, false, false);
            if (nt == 2) acc2 = __builtin_amdgcn_wmma_f32_16x16x32_f16(false, pa, false, bv, (short)0, acc2, false, false);
            if (nt == 3) acc3 = __builtin_amdgcn_wmma_f32_16x16x32_f16(false, pa, false, bv, (short)0, acc3, false, false);
        }
    }
    // spill each wave's 16x64 partial to LDS
#pragma unroll
    for (int r = 0; r < 8; ++r) {
        int row = r + 8 * hi;
        sAcc[wave * TQ * DIM + row * DIM +  0 + l16] = acc0[r];
        sAcc[wave * TQ * DIM + row * DIM + 16 + l16] = acc1[r];
        sAcc[wave * TQ * DIM + row * DIM + 32 + l16] = acc2[r];
        sAcc[wave * TQ * DIM + row * DIM + 48 + l16] = acc3[r];
    }
    __syncthreads();

    // ---- reduce across waves, store output (b128) ----
    {
        v4f* ov = (v4f*)(Out + (size_t)batch * SEQ * DIM + (size_t)qbase * DIM);
        for (int i = tid; i < TQ * DIM / 4; i += NT) {
            v4f s = (v4f){0.0f, 0.0f, 0.0f, 0.0f};
#pragma unroll
            for (int w = 0; w < NW; ++w) s += *(const v4f*)&sAcc[w * TQ * DIM + i * 4];
            ov[i] = s;
        }
    }
}

extern "C" void kernel_launch(void* const* d_in, const int* in_sizes, int n_in,
                              void* d_out, int out_size, void* d_ws, size_t ws_size,
                              hipStream_t stream) {
    const float* Q = (const float*)d_in[0];
    const float* K = (const float*)d_in[1];
    const float* V = (const float*)d_in[2];
    // d_in[3] is the mask flag (==1, causal) — baked into the kernel.
    float* Out  = (float*)d_out;
    float* Attn = Out + (size_t)BATCH * SEQ * DIM;

    dim3 grid(BATCH * (SEQ / TQ));   // 2048 workgroups, one 16-row q-tile each
    dim3 block(NT);
    attn_fused_kernel<<<grid, block, 0, stream>>>(Q, K, V, Out, Attn);
}
